// MoELayerwiseTransformerDecoder_51238959841832
// MI455X (gfx1250) — matrix operand order repaired and checked by reading, not compiled, not run on hardware
//
#include <hip/hip_runtime.h>

// ---------------------------------------------------------------------------
// MoE layerwise transformer decoder for MI455X (gfx1250, wave32, WMMA).
// All GEMM-class math: v_wmma_f32_16x16x32_bf16 (bf16 in, f32 acc).
// Operands pre-converted to bf16 once (halves global read traffic, removes
// cvt VALU from hot loops); GEMM tiles staged via GLOBAL_LOAD_ASYNC_TO_LDS_B128
// with double-buffered LDS and s_wait_asynccnt pipelining.
// Dims: L=4,E=4,D=512,DFF=2048,H=8,B=16,T=256,S=512,HD=64.
// ---------------------------------------------------------------------------

typedef __attribute__((ext_vector_type(16))) __bf16       v16bf;
typedef __attribute__((ext_vector_type(8)))  float        v8f;
typedef __attribute__((ext_vector_type(4)))  float        f32x4;
typedef __attribute__((ext_vector_type(4)))  unsigned int u32x4;
typedef __attribute__((ext_vector_type(2)))  unsigned int u32x2;

union Frag { v16bf v; u32x4 q[2]; unsigned short u[16]; };

__device__ __forceinline__ unsigned short f2bf(float f) {
    unsigned u = __float_as_uint(f);
    unsigned r = u + 0x7FFFu + ((u >> 16) & 1u);   // round-to-nearest-even
    return (unsigned short)(r >> 16);
}

__device__ __forceinline__ void store_bf4(unsigned short* p, f32x4 f) {
    unsigned lo = (unsigned)f2bf(f.x) | ((unsigned)f2bf(f.y) << 16);
    unsigned hi = (unsigned)f2bf(f.z) | ((unsigned)f2bf(f.w) << 16);
    u32x2 v; v.x = lo; v.y = hi;
    *reinterpret_cast<u32x2*>(p) = v;
}

// LDS byte offset of a generic pointer into __shared__ space.
__device__ __forceinline__ unsigned lds_off(const void* p) {
    return (unsigned)(unsigned long long)(const __attribute__((address_space(3))) void*)p;
}

// Async DMA: 16 bytes per lane, global -> LDS (tracked by ASYNCcnt).
__device__ __forceinline__ void async_b128(unsigned lds_byte,
                                           const unsigned short* gbase,
                                           unsigned goff_byte) {
    asm volatile("global_load_async_to_lds_b128 %0, %1, %2"
                 :: "v"(lds_byte), "v"(goff_byte), "s"(gbase)
                 : "memory");
}

// ---------------------------------------------------------------------------
// C[M,N] = act(A[M,K] @ W[N,K]^T + bias); A,W bf16 row-major, bias f32.
// OUTBF: write bf16 output (for operands feeding another GEMM/attention).
// Block: 256 thr = 8 waves; block tile 128x128; wave tile 64x32; K-step 32.
// Double-buffered LDS, async global->LDS copies one tile ahead.
// ---------------------------------------------------------------------------
template <bool RELU, bool OUTBF>
__global__ __launch_bounds__(256)
void k_gemm(const unsigned short* __restrict__ A, const unsigned short* __restrict__ W,
            const float* __restrict__ bias, void* __restrict__ Cout,
            int M, int N, int Kd)
{
    __shared__ alignas(16) unsigned short sA[2][128 * 40];
    __shared__ alignas(16) unsigned short sB[2][128 * 40];

    const int tid  = threadIdx.x;
    const int lane = tid & 31;
    const int wid  = tid >> 5;
    const int m0   = blockIdx.y * 128;
    const int n0   = blockIdx.x * 128;
    const int wm   = (wid >> 2) * 64;   // 0 / 64
    const int wn   = (wid & 3) * 32;    // 0..96
    const int lrow = lane & 15;
    const int cg   = lane >> 4;         // half-wave group
    const int ka   = cg * 8;            // A-frag element offset (ushorts)
    const int kb   = cg * 16;           // unused; B frag is also read at ka

    v8f zero = {};
    v8f acc[4][2];
#pragma unroll
    for (int i = 0; i < 4; i++) { acc[i][0] = zero; acc[i][1] = zero; }

    // staging map: 2 threads per row, 32B (16 bf16) each, as 2x b128 asyncs
    const int r_ld = tid >> 1;          // 0..127
    const int h_ld = (tid & 1) * 16;    // 0 / 16 (ushorts)

    auto issue = [&](int buf, int k0) {
        const unsigned short* ga = A + (size_t)m0 * Kd + k0;
        const unsigned short* gw = W + (size_t)n0 * Kd + k0;
        const unsigned go = (unsigned)(r_ld * Kd + h_ld) * 2u;
        const unsigned la = lds_off(&sA[buf][r_ld * 40 + h_ld]);
        const unsigned lb = lds_off(&sB[buf][r_ld * 40 + h_ld]);
        async_b128(la,      ga, go);
        async_b128(la + 16, ga, go + 16);
        async_b128(lb,      gw, go);
        async_b128(lb + 16, gw, go + 16);
    };

    const int nk = Kd >> 5;
    issue(0, 0);
    for (int kt = 0; kt < nk; ++kt) {
        if (kt + 1 < nk) {
            issue((kt + 1) & 1, (kt + 1) * 32);
            // in-order async completion: <=4 outstanding => tile kt landed
            asm volatile("s_wait_asynccnt 4" ::: "memory");
        } else {
            asm volatile("s_wait_asynccnt 0" ::: "memory");
        }
        __syncthreads();

        const unsigned short* cA = sA[kt & 1];
        const unsigned short* cB = sB[kt & 1];
        Frag bfr[2];
#pragma unroll
        for (int j = 0; j < 2; j++) {
            const unsigned short* src = &cB[(wn + j * 16 + lrow) * 40 + ka];
            bfr[j].q[0] = *reinterpret_cast<const u32x4*>(src);
            bfr[j].q[1] = *reinterpret_cast<const u32x4*>(src + 16);
        }
#pragma unroll
        for (int i = 0; i < 4; i++) {
            Frag af;
            const unsigned short* src = &cA[(wm + i * 16 + lrow) * 40 + ka];
            af.q[0] = *reinterpret_cast<const u32x4*>(src);
            af.q[1] = *reinterpret_cast<const u32x4*>(src + 16);
            acc[i][0] = __builtin_amdgcn_wmma_f32_16x16x32_bf16(
                false, af.v, false, bfr[0].v, (short)0, acc[i][0], false, false);
            acc[i][1] = __builtin_amdgcn_wmma_f32_16x16x32_bf16(
                false, af.v, false, bfr[1].v, (short)0, acc[i][1], false, false);
        }
        __syncthreads();
    }

    float*          Cf = (float*)Cout;
    unsigned short* Cb = (unsigned short*)Cout;
#pragma unroll
    for (int i = 0; i < 4; i++) {
        const int rowb = m0 + wm + i * 16 + cg * 8;
#pragma unroll
        for (int j = 0; j < 2; j++) {
            const int col = n0 + wn + j * 16 + lrow;
            const float bb = bias[col];
#pragma unroll
            for (int r = 0; r < 8; r++) {
                float v = acc[i][j][r] + bb;
                if (RELU) v = fmaxf(v, 0.0f);
                const size_t idx = (size_t)(rowb + r) * N + col;
                if (OUTBF) Cb[idx] = f2bf(v);
                else       Cf[idx] = v;
            }
        }
    }
}

// ---------------------------------------------------------------------------
// Flash attention on bf16 Q/K/V: block = 4 waves, each wave owns 16 Q rows;
// waves share K (async-copied) / V (transposed) chunks of 32 keys in LDS.
// Online softmax per wave; bf16 output (feeds out-proj GEMM).
// ---------------------------------------------------------------------------
__global__ __launch_bounds__(128)
void k_attn(const unsigned short* __restrict__ Qb, int q_bs, int q_rs, int q_co,
            const unsigned short* __restrict__ Kb, int k_bs, int k_rs, int k_co,
            const unsigned short* __restrict__ Vb, int v_bs, int v_rs, int v_co,
            unsigned short* __restrict__ O, int S, int T, float scale)
{
    __shared__ alignas(16) unsigned short sK[32 * 72];     // [s][hd]
    __shared__ alignas(16) unsigned short sV[64 * 40];     // [hd][s] (transposed)
    __shared__ alignas(16) unsigned short sP[4][16 * 40];  // per-wave P tile

    const int b    = blockIdx.z, h = blockIdx.y;
    const int tid  = threadIdx.x;
    const int lane = tid & 31;
    const int wid  = tid >> 5;
    const int lrow = lane & 15;
    const int cg   = lane >> 4;
    const int ka   = cg * 8;
    const int kb   = cg * 16;
    const int qr0  = blockIdx.x * 64 + wid * 16;

    const unsigned short* Q  = Qb + (size_t)b * q_bs + q_co + h * 64;
    const unsigned short* Kp = Kb + (size_t)b * k_bs + k_co + h * 64;
    const unsigned short* Vp = Vb + (size_t)b * v_bs + v_co + h * 64;

    // Q fragments (A operand), loaded once straight from global bf16.
    Frag qf[2];
    {
        const unsigned short* qr = Q + (size_t)(qr0 + lrow) * q_rs;
#pragma unroll
        for (int f = 0; f < 2; f++) {
            qf[f].q[0] = *reinterpret_cast<const u32x4*>(qr + f * 32 + ka);
            qf[f].q[1] = *reinterpret_cast<const u32x4*>(qr + f * 32 + ka + 16);
        }
    }

    float mrow[8], lsum[8];
    v8f zero = {};
    v8f accO[4];
#pragma unroll
    for (int r = 0; r < 8; r++) { mrow[r] = -3.0e38f; lsum[r] = 0.0f; }
#pragma unroll
    for (int c = 0; c < 4; c++) accO[c] = zero;

    // staging map: 4 threads per key row; 32B K via asyncs, 16 bf16 V scatter
    const int r_st = tid >> 2;          // 0..31 (key row)
    const int c_st = (tid & 3) * 16;    // hd offset (ushorts)

    const int nch = S >> 5;
    for (int ch = 0; ch < nch; ++ch) {
        const int s0 = ch * 32;
        {
            const unsigned go = (unsigned)((s0 + r_st) * k_rs + c_st) * 2u;
            const unsigned lk = lds_off(&sK[r_st * 72 + c_st]);
            async_b128(lk,      Kp, go);
            async_b128(lk + 16, Kp, go + 16);

            const unsigned short* vp = Vp + (size_t)(s0 + r_st) * v_rs + c_st;
            union { u32x4 q; unsigned short u[8]; } t0, t1;
            t0.q = *reinterpret_cast<const u32x4*>(vp);
            t1.q = *reinterpret_cast<const u32x4*>(vp + 8);
#pragma unroll
            for (int i = 0; i < 8; i++) sV[(c_st + i) * 40 + r_st]     = t0.u[i];
#pragma unroll
            for (int i = 0; i < 8; i++) sV[(c_st + 8 + i) * 40 + r_st] = t1.u[i];
        }
        asm volatile("s_wait_asynccnt 0" ::: "memory");
        __syncthreads();

        // scores: 16x32 tile = two 16x16 WMMA outputs, K=64 in two steps
        v8f sc[2]; sc[0] = zero; sc[1] = zero;
#pragma unroll
        for (int j = 0; j < 2; j++) {
#pragma unroll
            for (int f = 0; f < 2; f++) {
                Frag kf;
                const unsigned short* src = &sK[(j * 16 + lrow) * 72 + f * 32 + ka];
                kf.q[0] = *reinterpret_cast<const u32x4*>(src);
                kf.q[1] = *reinterpret_cast<const u32x4*>(src + 16);
                sc[j] = __builtin_amdgcn_wmma_f32_16x16x32_bf16(
                    false, qf[f].v, false, kf.v, (short)0, sc[j], false, false);
            }
        }

        // online softmax (rows live per half-wave; reduce across 16 lanes)
        float p0[8], p1[8];
#pragma unroll
        for (int r = 0; r < 8; r++) {
            const float s0v = sc[0][r] * scale;
            const float s1v = sc[1][r] * scale;
            float mx = fmaxf(s0v, s1v);
            mx = fmaxf(mx, __shfl_xor(mx, 1));
            mx = fmaxf(mx, __shfl_xor(mx, 2));
            mx = fmaxf(mx, __shfl_xor(mx, 4));
            mx = fmaxf(mx, __shfl_xor(mx, 8));
            const float mnew = fmaxf(mrow[r], mx);
            const float corr = __expf(mrow[r] - mnew);
            mrow[r] = mnew;
            p0[r] = __expf(s0v - mnew);
            p1[r] = __expf(s1v - mnew);
            float rs = p0[r] + p1[r];
            rs += __shfl_xor(rs, 1);
            rs += __shfl_xor(rs, 2);
            rs += __shfl_xor(rs, 4);
            rs += __shfl_xor(rs, 8);
            lsum[r] = lsum[r] * corr + rs;
            accO[0][r] *= corr; accO[1][r] *= corr;
            accO[2][r] *= corr; accO[3][r] *= corr;
        }

        // reshape P: C-layout -> A-layout via per-wave LDS tile
        unsigned short* pw = &sP[wid][0];
#pragma unroll
        for (int r = 0; r < 8; r++) {
            pw[(r + cg * 8) * 40 + lrow]      = f2bf(p0[r]);
            pw[(r + cg * 8) * 40 + 16 + lrow] = f2bf(p1[r]);
        }
        Frag pf;
        {
            const unsigned short* src = &sP[wid][lrow * 40 + ka];
            pf.q[0] = *reinterpret_cast<const u32x4*>(src);
            pf.q[1] = *reinterpret_cast<const u32x4*>(src + 16);
        }
        // O += P @ V  (V^T staged: B-fragment rows are hd)
#pragma unroll
        for (int c = 0; c < 4; c++) {
            Frag vf;
            const unsigned short* src = &sV[(c * 16 + lrow) * 40 + ka];
            vf.q[0] = *reinterpret_cast<const u32x4*>(src);
            vf.q[1] = *reinterpret_cast<const u32x4*>(src + 16);
            accO[c] = __builtin_amdgcn_wmma_f32_16x16x32_bf16(
                false, pf.v, false, vf.v, (short)0, accO[c], false, false);
        }
        __syncthreads();
    }

#pragma unroll
    for (int r = 0; r < 8; r++) {
        const float inv = 1.0f / lsum[r];
        const int t = qr0 + r + cg * 8;
        unsigned short* orow = O + ((size_t)b * T + t) * 512 + h * 64;
#pragma unroll
        for (int c = 0; c < 4; c++)
            orow[c * 16 + lrow] = f2bf(accO[c][r] * inv);
    }
}

// ---------------------------------------------------------------------------
// Router: pooled mean over T, logits, top-2 softmax -> dense wf[B,E]
// ---------------------------------------------------------------------------
__global__ __launch_bounds__(256)
void k_router(const float* __restrict__ x, const float* __restrict__ rw,
              float* __restrict__ wf)
{
    __shared__ float pooled[512];
    __shared__ float logits[4];
    const int b = blockIdx.x;
    for (int d = threadIdx.x; d < 512; d += 256) {
        float s = 0.0f;
        for (int t = 0; t < 256; t++) s += x[((size_t)b * 256 + t) * 512 + d];
        pooled[d] = s * (1.0f / 256.0f);
    }
    __syncthreads();
    if (threadIdx.x < 4) {
        const int e = threadIdx.x;
        float s = 0.0f;
        for (int d = 0; d < 512; d++) s += pooled[d] * rw[e * 512 + d];
        logits[e] = s;
    }
    __syncthreads();
    if (threadIdx.x == 0) {
        int i0 = 0;
        for (int e = 1; e < 4; e++) if (logits[e] > logits[i0]) i0 = e;
        int i1 = -1;
        for (int e = 0; e < 4; e++) {
            if (e == i0) continue;
            if (i1 < 0 || logits[e] > logits[i1]) i1 = e;
        }
        const float a = logits[i0], c = logits[i1];
        const float ea = 1.0f, ec = __expf(c - a);
        const float inv = 1.0f / (ea + ec);
        for (int e = 0; e < 4; e++) wf[b * 4 + e] = 0.0f;
        wf[b * 4 + i0] = ea * inv;
        wf[b * 4 + i1] = ec * inv;
    }
}

// ---------------------------------------------------------------------------
// LayerNorm(xa + xb) per row (wave-per-row, D=512). mode: 0 -> out = y
// (+ optional bf16 copy); 1 -> out = wf[b,e]*y; 2 -> out += wf[b,e]*y.
// ---------------------------------------------------------------------------
__global__ __launch_bounds__(256)
void k_ln(const float* __restrict__ xa, const float* __restrict__ xb,
          const float* __restrict__ gamma, const float* __restrict__ beta,
          float* __restrict__ out, unsigned short* __restrict__ out_bf,
          const float* __restrict__ wf, int eidx, int mode, int nrows)
{
    const int row = blockIdx.x * 8 + (threadIdx.x >> 5);
    if (row >= nrows) return;
    const int lane = threadIdx.x & 31;
    const float* pa = xa + (size_t)row * 512;
    const float* pb = xb + (size_t)row * 512;
    float v[16];
    float s = 0.0f;
#pragma unroll
    for (int i = 0; i < 16; i++) { const int d = lane + i * 32; v[i] = pa[d] + pb[d]; s += v[i]; }
    for (int m = 1; m < 32; m <<= 1) s += __shfl_xor(s, m);
    const float mean = s * (1.0f / 512.0f);
    float var = 0.0f;
#pragma unroll
    for (int i = 0; i < 16; i++) { const float d = v[i] - mean; var += d * d; }
    for (int m = 1; m < 32; m <<= 1) var += __shfl_xor(var, m);
    const float inv = rsqrtf(var * (1.0f / 512.0f) + 1e-5f);
    float scl = 1.0f; bool accfl = false;
    if (mode) { scl = wf[(row >> 8) * 4 + eidx]; accfl = (mode == 2); }
    float* po = out + (size_t)row * 512;
    unsigned short* pq = out_bf ? out_bf + (size_t)row * 512 : nullptr;
#pragma unroll
    for (int i = 0; i < 16; i++) {
        const int d = lane + i * 32;
        const float r = (v[i] - mean) * inv * gamma[d] + beta[d];
        if (mode == 0) {
            po[d] = r;
            if (pq) pq[d] = f2bf(r);
        } else {
            float o = scl * r;
            if (accfl) o += po[d];
            po[d] = o;
        }
    }
}

__global__ void k_cvt(const float* __restrict__ a, unsigned short* __restrict__ o, long n4)
{
    long i = (long)blockIdx.x * blockDim.x + threadIdx.x;
    const long stride = (long)gridDim.x * blockDim.x;
    for (; i < n4; i += stride)
        store_bf4(o + i * 4, reinterpret_cast<const f32x4*>(a)[i]);
}

__global__ void k_copy(const float* __restrict__ a, float* __restrict__ b, long n)
{
    long i = (long)blockIdx.x * blockDim.x + threadIdx.x;
    const long stride = (long)gridDim.x * blockDim.x;
    for (; i < n; i += stride) b[i] = a[i];
}

// ---------------------------------------------------------------------------
// Host orchestration
// ---------------------------------------------------------------------------
extern "C" void kernel_launch(void* const* d_in, const int* in_sizes, int n_in,
                              void* d_out, int out_size, void* d_ws, size_t ws_size,
                              hipStream_t stream)
{
    (void)in_sizes; (void)n_in; (void)out_size; (void)ws_size;

    const float* tgt      = (const float*)d_in[0];
    const float* memory_  = (const float*)d_in[1];
    const float* router_w = (const float*)d_in[2];
    const float* sa_in_w  = (const float*)d_in[3];
    const float* sa_in_b  = (const float*)d_in[4];
    const float* sa_out_w = (const float*)d_in[5];
    const float* sa_out_b = (const float*)d_in[6];
    const float* ca_in_w  = (const float*)d_in[7];
    const float* ca_in_b  = (const float*)d_in[8];
    const float* ca_out_w = (const float*)d_in[9];
    const float* ca_out_b = (const float*)d_in[10];
    const float* lin1_w   = (const float*)d_in[11];
    const float* lin1_b   = (const float*)d_in[12];
    const float* lin2_w   = (const float*)d_in[13];
    const float* lin2_b   = (const float*)d_in[14];
    const float* n1_g     = (const float*)d_in[15];
    const float* n1_b     = (const float*)d_in[16];
    const float* n2_g     = (const float*)d_in[17];
    const float* n2_b     = (const float*)d_in[18];
    const float* n3_g     = (const float*)d_in[19];
    const float* n3_b     = (const float*)d_in[20];

    constexpr int Bn = 16, Tn = 256, Sn = 512, Dn = 512, DFFn = 2048, Ln = 4, En = 4, Hn = 8;
    const size_t BT = (size_t)Bn * Tn;   // 4096
    const size_t BS = (size_t)Bn * Sn;   // 8192

    // bump allocator over workspace, 256B granularity
    char* wsb = (char*)d_ws;
    size_t off = 0;
    auto allocf = [&](size_t n) { float* p = (float*)(wsb + off); off += ((n * 4 + 255) & ~(size_t)255); return p; };
    auto allocu = [&](size_t n) { unsigned short* p = (unsigned short*)(wsb + off); off += ((n * 2 + 255) & ~(size_t)255); return p; };

    float* bufA = allocf(BT * Dn);
    float* bufB = allocf(BT * Dn);
    float* wf   = allocf((size_t)Bn * En);
    float* prj  = allocf(BT * Dn);
    float* x2   = allocf(BT * Dn);
    float* x3   = allocf(BT * Dn);
    float* ff2  = allocf(BT * Dn);
    unsigned short* cur_bf = allocu(BT * Dn);
    unsigned short* mem_bf = allocu(BS * Dn);
    unsigned short* qkv_bf = allocu(BT * 3 * Dn);
    unsigned short* qcx_bf = allocu(BT * Dn);
    unsigned short* kvx_bf = allocu(BS * 2 * Dn);
    unsigned short* att_bf = allocu(BT * Dn);
    unsigned short* x2_bf  = allocu(BT * Dn);
    unsigned short* x3_bf  = allocu(BT * Dn);
    unsigned short* ff1_bf = allocu(BT * DFFn);
    unsigned short* wbA    = allocu((size_t)3 * Dn * Dn);   // qkv in-proj weights
    unsigned short* wbB    = allocu((size_t)Dn * Dn);       // out-proj weights
    unsigned short* wbC    = allocu((size_t)DFFn * Dn);     // ffn weights

    k_copy<<<512, 256, 0, stream>>>(tgt, bufA, (long)(BT * Dn));
    k_cvt<<<1024, 256, 0, stream>>>(memory_, mem_bf, (long)(BS * Dn / 4));
    float* cur = bufA; float* nxt = bufB;

    for (int l = 0; l < Ln; ++l) {
        k_router<<<Bn, 256, 0, stream>>>(cur, router_w + (size_t)l * En * Dn, wf);
        k_cvt<<<1024, 256, 0, stream>>>(cur, cur_bf, (long)(BT * Dn / 4));
        for (int e = 0; e < En; ++e) {
            const size_t le = (size_t)l * En + e;
            const float* w_sain  = sa_in_w  + le * 3 * Dn * Dn;
            const float* b_sain  = sa_in_b  + le * 3 * Dn;
            const float* w_saout = sa_out_w + le * Dn * Dn;
            const float* b_saout = sa_out_b + le * Dn;
            const float* w_cain  = ca_in_w  + le * 3 * Dn * Dn;
            const float* b_cain  = ca_in_b  + le * 3 * Dn;
            const float* w_caout = ca_out_w + le * Dn * Dn;
            const float* b_caout = ca_out_b + le * Dn;
            const float* w_l1 = lin1_w + le * (size_t)DFFn * Dn;
            const float* b_l1 = lin1_b + le * DFFn;
            const float* w_l2 = lin2_w + le * (size_t)Dn * DFFn;
            const float* b_l2 = lin2_b + le * Dn;
            const float* g1 = n1_g + le * Dn; const float* be1 = n1_b + le * Dn;
            const float* g2 = n2_g + le * Dn; const float* be2 = n2_b + le * Dn;
            const float* g3 = n3_g + le * Dn; const float* be3 = n3_b + le * Dn;

            // ---- self attention ----
            k_cvt<<<1024, 256, 0, stream>>>(w_sain, wbA, (long)(3 * Dn * Dn / 4));
            k_gemm<false, true><<<dim3(12, 32), 256, 0, stream>>>(
                cur_bf, wbA, b_sain, qkv_bf, (int)BT, 3 * Dn, Dn);
            k_attn<<<dim3(Tn / 64, Hn, Bn), 128, 0, stream>>>(
                qkv_bf, Tn * 3 * Dn, 3 * Dn, 0,
                qkv_bf, Tn * 3 * Dn, 3 * Dn, Dn,
                qkv_bf, Tn * 3 * Dn, 3 * Dn, 2 * Dn,
                att_bf, Tn, Tn, 0.125f);
            k_cvt<<<1024, 256, 0, stream>>>(w_saout, wbB, (long)(Dn * Dn / 4));
            k_gemm<false, false><<<dim3(4, 32), 256, 0, stream>>>(
                att_bf, wbB, b_saout, prj, (int)BT, Dn, Dn);
            k_ln<<<512, 256, 0, stream>>>(cur, prj, g1, be1, x2, x2_bf, nullptr, 0, 0, (int)BT);

            // ---- cross attention ----
            k_cvt<<<1024, 256, 0, stream>>>(w_cain, wbA, (long)(3 * Dn * Dn / 4));
            k_gemm<false, true><<<dim3(4, 32), 256, 0, stream>>>(
                x2_bf, wbA, b_cain, qcx_bf, (int)BT, Dn, Dn);
            k_gemm<false, true><<<dim3(8, 64), 256, 0, stream>>>(
                mem_bf, wbA + (size_t)Dn * Dn, b_cain + Dn, kvx_bf, (int)BS, 2 * Dn, Dn);
            k_attn<<<dim3(Tn / 64, Hn, Bn), 128, 0, stream>>>(
                qcx_bf, Tn * Dn, Dn, 0,
                kvx_bf, Sn * 2 * Dn, 2 * Dn, 0,
                kvx_bf, Sn * 2 * Dn, 2 * Dn, Dn,
                att_bf, Sn, Tn, 0.125f);
            k_cvt<<<1024, 256, 0, stream>>>(w_caout, wbB, (long)(Dn * Dn / 4));
            k_gemm<false, false><<<dim3(4, 32), 256, 0, stream>>>(
                att_bf, wbB, b_caout, prj, (int)BT, Dn, Dn);
            k_ln<<<512, 256, 0, stream>>>(x2, prj, g2, be2, x3, x3_bf, nullptr, 0, 0, (int)BT);

            // ---- FFN ----
            k_cvt<<<1024, 256, 0, stream>>>(w_l1, wbC, (long)(DFFn * Dn / 4));
            k_gemm<true, true><<<dim3(16, 32), 256, 0, stream>>>(
                x3_bf, wbC, b_l1, ff1_bf, (int)BT, DFFn, Dn);
            k_cvt<<<1024, 256, 0, stream>>>(w_l2, wbC, (long)(Dn * DFFn / 4));
            k_gemm<false, false><<<dim3(4, 32), 256, 0, stream>>>(
                ff1_bf, wbC, b_l2, ff2, (int)BT, Dn, DFFn);
            // LN3 fused with MoE combine: nxt (+)= wf[b,e] * LN(x3 + ffn)
            k_ln<<<512, 256, 0, stream>>>(x3, ff2, g3, be3, nxt, nullptr, wf, e,
                                          (e == 0) ? 1 : 2, (int)BT);
        }
        float* t = cur; cur = nxt; nxt = t;
    }
    k_copy<<<512, 256, 0, stream>>>(cur, (float*)d_out, (long)(BT * Dn));
}
